// gtnet_64768106824359
// MI455X (gfx1250) — compile-verified, matrix operand order checked by reading, not run on hardware
//
#include <hip/hip_runtime.h>
#include <cstdint>

typedef __attribute__((ext_vector_type(16))) _Float16 v16h;
typedef __attribute__((ext_vector_type(8)))  float    v8f;

#define B_  16
#define N_  1000

__device__ __forceinline__ float sigmoidf_(float x) { return 1.f / (1.f + __expf(-x)); }

// ============================ WMMA GEMM (templated, branch-free) ============================
// C[M,N] = op(A)[M,K] * op(B)[K,N] (+ bias[n]); A row-major (lda), optional compile-time relu(A).
// BTRANS==false: B[k*ldb+n] ; BTRANS==true: B[n*ldb+k]. One wave per 16x16 tile.
// Requires M%16==0, N%16==0, K in {16, 32, 48, 64, ...} (multiple of 16).
// fp32 in/out, f16 WMMA operands, f32 accumulate. EXEC stays all-ones (no divergence).

template <bool RELU>
__device__ __forceinline__ void load8h(v16h& v, int base, const float* __restrict__ p)
{
    float4 a = *(const float4*)p;
    float4 b = *(const float4*)(p + 4);
    float t[8] = {a.x, a.y, a.z, a.w, b.x, b.y, b.z, b.w};
#pragma unroll
    for (int e = 0; e < 8; ++e) {
        float x = t[e];
        if (RELU) x = fmaxf(x, 0.f);
        v[base + e] = (_Float16)x;
    }
}
__device__ __forceinline__ void zero8h(v16h& v, int base)
{
#pragma unroll
    for (int e = 0; e < 8; ++e) v[base + e] = (_Float16)0.f;
}

template <int K, bool BTRANS, bool RELU>
__global__ __launch_bounds__(32)
void wmma_gemm_t(const float* __restrict__ A, int lda,
                 const float* __restrict__ B, int ldb,
                 float* __restrict__ C, int ldc,
                 int tilesN, const float* __restrict__ bias)
{
    int tile = blockIdx.x;
    int tm = tile / tilesN, tn = tile - tm * tilesN;
    int m0 = tm << 4, n0 = tn << 4;
    int lane = threadIdx.x;
    int lr = lane & 15, lh = lane >> 4;

    v8f acc = {};
    if (bias) {                       // uniform branch
        float bv = bias[n0 + lr];
#pragma unroll
        for (int r = 0; r < 8; ++r) acc[r] = bv;
    }
    const float* Arow = A + (long long)(m0 + lr) * lda;
    const float* Brow = BTRANS ? (B + (long long)(n0 + lr) * ldb) : (B + (n0 + lr));

#pragma unroll
    for (int k0 = 0; k0 < K; k0 += 32) {
        v16h av, bvv;
        // A, first half: k = k0 + 8*lh + e (e=0..7); always valid for existing chunk
        load8h<RELU>(av, 0, Arow + k0 + 8 * lh);
        // A, second half: k = k0 + 16 + 8*lh + (e-8); valid iff chunk is full
        if (k0 + 32 <= K) load8h<RELU>(av, 8, Arow + k0 + 16 + 8 * lh);
        else              zero8h(av, 8);

        if (BTRANS) {
            load8h<false>(bvv, 0, Brow + k0 + 8 * lh);
            if (k0 + 32 <= K) load8h<false>(bvv, 8, Brow + k0 + 16 + 8 * lh);
            else              zero8h(bvv, 8);
        } else {
#pragma unroll
            for (int e = 0; e < 8; ++e)
                bvv[e] = (_Float16)Brow[(long long)(k0 + 8 * lh + e) * ldb];
            if (k0 + 32 <= K) {
#pragma unroll
                for (int e = 0; e < 8; ++e)
                    bvv[8 + e] = (_Float16)Brow[(long long)(k0 + 16 + 8 * lh + e) * ldb];
            } else {
                zero8h(bvv, 8);
            }
        }
        acc = __builtin_amdgcn_wmma_f32_16x16x32_f16(false, av, false, bvv,
                                                     (short)0, acc, false, false);
    }
#pragma unroll
    for (int r = 0; r < 8; ++r) {  // C/D: VGPR r -> M = 8*lh + r, N = lane&15
        int m = m0 + (lh << 3) + r;
        C[(long long)m * ldc + n0 + lr] = acc[r];
    }
}

// ============================ start conv ============================
// h[b,o,n,t] = sum_c x[b,c,n,t] * W[c,o] + b[o];   h: (B,32,N,64)
__global__ void k_start(const float* __restrict__ x, const float* __restrict__ W,
                        const float* __restrict__ bb, float* __restrict__ h)
{
    long long i = blockIdx.x * (long long)blockDim.x + threadIdx.x;
    long long total = (long long)B_ * 32 * N_ * 64;
    if (i >= total) return;
    int t = (int)(i & 63); long long r = i >> 6;
    int n = (int)(r % N_); r /= N_;
    int o = (int)(r % 32); int b = (int)(r / 32);
    float s = bb[o];
#pragma unroll
    for (int c = 0; c < 12; ++c)
        s += x[(((long long)b * 12 + c) * N_ + n) * 64 + t] * W[c * 32 + o];
    h[i] = s;
}

// ============================ skip conv ============================
// skip(b,n,o) (+)= sum_{c,t} hin[b,c,n,t] * W[o,c,t] + bias[o];  skip layout (B,N,64)
__global__ void k_skip_conv(const float* __restrict__ hin, const float* __restrict__ W,
                            const float* __restrict__ bias, float* __restrict__ skip,
                            int C, int T, int accumulate)
{
    long long i = blockIdx.x * (long long)blockDim.x + threadIdx.x;
    if (i >= (long long)B_ * N_ * 64) return;
    int o = (int)(i & 63); long long r = i >> 6;
    int n = (int)(r % N_); int b = (int)(r / N_);
    float s = bias[o];
    for (int c = 0; c < C; ++c) {
        const float* hp = hin + (((long long)b * C + c) * N_ + n) * T;
        const float* wp = W + ((long long)o * C + c) * T;
        __builtin_prefetch(wp + T, 0, 1);   // global_prefetch_b8 next weight row
        float acc = 0.f;
        for (int t = 0; t < T; ++t) acc += hp[t] * wp[t];
        s += acc;
    }
    if (accumulate) skip[i] += s; else skip[i] = s;
}

// ============================ dilated inception (fused tanh*sigmoid gate) ============================
struct IncArgs {
    const float* fw[4]; const float* fb[4];
    const float* gw[4]; const float* gb[4];
};
__global__ void k_inception(const float* __restrict__ hin, float* __restrict__ out,
                            IncArgs wts, int Tin, int Tout)
{
    long long i = blockIdx.x * (long long)blockDim.x + threadIdx.x;
    long long total = (long long)B_ * 32 * N_ * Tout;
    if (i >= total) return;
    int t = (int)(i % Tout); long long r = i / Tout;
    int n = (int)(r % N_); r /= N_;
    int oc = (int)(r % 32); int b = (int)(r / 32);
    int br = oc >> 3, occ = oc & 7;
    int k = (br == 0) ? 2 : (br == 1) ? 3 : (br == 2) ? 6 : 7;
    int tstart = t + (Tin - k + 1 - Tout);   // truncate branch to last Tout steps
    const float* Wf = wts.fw[br] + (long long)occ * 32 * k;
    const float* Wg = wts.gw[br] + (long long)occ * 32 * k;
    float fs = wts.fb[br][occ], gs = wts.gb[br][occ];
    for (int c = 0; c < 32; ++c) {
        const float* hp = hin + (((long long)b * 32 + c) * N_ + n) * Tin + tstart;
        for (int kk = 0; kk < k; ++kk) {
            float hv = hp[kk];
            fs += hv * Wf[c * k + kk];
            gs += hv * Wg[c * k + kk];
        }
    }
    out[i] = tanhf(fs) * sigmoidf_(gs);
}

// ============================ residual tail add ============================
__global__ void k_res_add(float* __restrict__ tfout, const float* __restrict__ resid,
                          int Tin, int Tout)
{
    long long i = blockIdx.x * (long long)blockDim.x + threadIdx.x;
    long long total = (long long)B_ * 32 * N_ * Tout;
    if (i >= total) return;
    int t = (int)(i % Tout); long long r = i / Tout;
    int n = (int)(r % N_); r /= N_;
    int c = (int)(r % 32); int b = (int)(r / 32);
    tfout[i] += resid[(((long long)b * 32 + c) * N_ + n) * Tin + t + (Tin - Tout)];
}

// ============================ transformer attention (tiny: E <= 58 edges) ============================
__global__ void k_attn(const float* __restrict__ xf, float* __restrict__ out,
                       const int* __restrict__ src, const int* __restrict__ dst, int E,
                       const float* __restrict__ Wq, const float* __restrict__ bq,
                       const float* __restrict__ Wk, const float* __restrict__ bk,
                       const float* __restrict__ Wv, const float* __restrict__ bv)
{
    __shared__ float s_alpha[64];
    __shared__ int   s_dst[64];
    int e = threadIdx.x;
    if (e < E) {
        int sN = src[e], dN = dst[e];
        s_dst[e] = dN;
        float al = 0.f;
        for (int j = 0; j < 16; ++j) {
            float aq = bq[j], ak = bk[j];
            for (int c = 0; c < 16; ++c) {
                aq += xf[(long long)dN * 16 + c] * Wq[c * 16 + j];
                ak += xf[(long long)sN * 16 + c] * Wk[c * 16 + j];
            }
            al += aq * ak;
        }
        s_alpha[e] = al * 0.25f;  // / sqrt(16)
    }
    __syncthreads();
    if (e < E) {
        int dN = s_dst[e];
        float amax = -3.4e38f;
        for (int e2 = 0; e2 < E; ++e2) if (s_dst[e2] == dN) amax = fmaxf(amax, s_alpha[e2]);
        float den = 0.f;
        for (int e2 = 0; e2 < E; ++e2) if (s_dst[e2] == dN) den += __expf(s_alpha[e2] - amax);
        float w = __expf(s_alpha[e] - amax) / den;
        int sN = src[e];
        for (int j = 0; j < 16; ++j) {
            float v = bv[j];
            for (int c = 0; c < 16; ++c) v += xf[(long long)sN * 16 + c] * Wv[c * 16 + j];
            atomicAdd(&out[(long long)dN * 16 + j], w * v);
        }
    }
}

// ============================ layer norm (per batch over C,N,T) ============================
__global__ void k_ln_reduce(const float* __restrict__ x, float* __restrict__ stats, int perB)
{
    int b = blockIdx.y;
    long long base = (long long)b * perB;
    float s = 0.f, s2 = 0.f;
    for (long long i = blockIdx.x * (long long)blockDim.x + threadIdx.x; i < perB;
         i += (long long)gridDim.x * blockDim.x) {
        float v = x[base + i]; s += v; s2 += v * v;
    }
    __shared__ float sh0[256], sh1[256];
    sh0[threadIdx.x] = s; sh1[threadIdx.x] = s2;
    __syncthreads();
    for (int off = 128; off; off >>= 1) {
        if ((int)threadIdx.x < off) { sh0[threadIdx.x] += sh0[threadIdx.x + off];
                                      sh1[threadIdx.x] += sh1[threadIdx.x + off]; }
        __syncthreads();
    }
    if (threadIdx.x == 0) { atomicAdd(&stats[2 * b], sh0[0]); atomicAdd(&stats[2 * b + 1], sh1[0]); }
}

__global__ void k_ln_apply(float* __restrict__ x, const float* __restrict__ stats,
                           const float* __restrict__ w, const float* __restrict__ bp, int perB)
{
    long long i = blockIdx.x * (long long)blockDim.x + threadIdx.x;
    long long total = (long long)B_ * perB;
    if (i >= total) return;
    int b = (int)(i / perB); long long r = i % perB;
    float inv = 1.f / (float)perB;
    float mu = stats[2 * b] * inv;
    float var = stats[2 * b + 1] * inv - mu * mu;
    x[i] = (x[i] - mu) * rsqrtf(var + 1e-5f) * w[r] + bp[r];
}

// ============================ small utility kernels ============================
__global__ void k_fill(float* __restrict__ p, float v, long long n)
{
    long long i = blockIdx.x * (long long)blockDim.x + threadIdx.x;
    if (i < n) p[i] = v;
}
__global__ void k_copy(const float* __restrict__ s, float* __restrict__ d, long long n)
{
    long long i = blockIdx.x * (long long)blockDim.x + threadIdx.x;
    if (i < n) d[i] = s[i];
}
// (B,N,128) -> (B,128,N)
__global__ void k_transpose_y(const float* __restrict__ ymat, float* __restrict__ yflat)
{
    long long i = blockIdx.x * (long long)blockDim.x + threadIdx.x;
    if (i >= (long long)B_ * 128 * N_) return;
    int n = (int)(i % N_); long long r = i / N_;
    int o = (int)(r % 128); int b = (int)(r / 128);
    yflat[i] = ymat[((long long)b * N_ + n) * 128 + o];
}

// ============================ GGC max-aggregation ============================
__global__ void k_ggc_aggr(const float* __restrict__ m, const int* __restrict__ src,
                           const int* __restrict__ dst, float* __restrict__ agg, int E)
{
    int d = threadIdx.x;  // 0..63, each thread owns one feature dim (race-free)
    for (int e = 0; e < E; ++e) {
        int s = src[e], j = dst[e];
        float v = m[(long long)s * 64 + d];
        float* a = &agg[(long long)j * 64 + d];
        if (v > *a) *a = v;
    }
}
__global__ void k_ggc_fix(float* __restrict__ agg, long long n)
{
    long long i = blockIdx.x * (long long)blockDim.x + threadIdx.x;
    if (i < n && agg[i] < -1e37f) agg[i] = 0.f;  // no in-edges -> 0
}

// ============================ GRU cell ============================
__global__ void k_gru(float* __restrict__ h, const float* __restrict__ gi,
                      const float* __restrict__ gh, long long Nn)
{
    long long i = blockIdx.x * (long long)blockDim.x + threadIdx.x;
    if (i >= Nn * 64) return;
    long long node = i >> 6; int j = (int)(i & 63);
    long long bse = node * 192;
    float r = sigmoidf_(gi[bse + j] + gh[bse + j]);
    float z = sigmoidf_(gi[bse + 64 + j] + gh[bse + 64 + j]);
    float g = tanhf(gi[bse + 128 + j] + r * gh[bse + 128 + j]);
    h[i] = (1.f - z) * g + z * h[i];
}

// ============================ end_conv_2 (fused y+yf, relu, 128->12) ============================
__global__ void k_e2(const float* __restrict__ yflat, const float* __restrict__ hg,
                     const float* __restrict__ W, const float* __restrict__ bias,
                     float* __restrict__ out)
{
    long long i = blockIdx.x * (long long)blockDim.x + threadIdx.x;
    if (i >= (long long)B_ * 12 * N_) return;
    int n = (int)(i % N_); long long r = i / N_;
    int o = (int)(r % 12); int b = (int)(r / 12);
    float s = bias[o];
    for (int c = 0; c < 128; ++c) {
        long long f = ((long long)b * 128 + c) * N_ + n;
        float v = yflat[f] + hg[f];
        v = v > 0.f ? v : 0.f;
        s += v * W[c * 12 + o];
    }
    out[i] = s;
}

// ============================ host orchestration ============================
extern "C" void kernel_launch(void* const* d_in, const int* in_sizes, int n_in,
                              void* d_out, int out_size, void* d_ws, size_t ws_size,
                              hipStream_t stream)
{
    (void)in_sizes; (void)n_in; (void)out_size; (void)ws_size;
    const float* x   = (const float*)d_in[0];
    const int*   edg = (const int*)d_in[1];
    const int*   src = edg;          // edg[0,:]  (1024)
    const int*   dst = edg + 1024;   // edg[1,:]
    auto F = [&](int i) { return (const float*)d_in[i]; };

    const float* start_W = F(2);  const float* start_b = F(3);
    const float* skip0_W = F(4);  const float* skip0_b = F(5);
    // layers: base = 6 + L*28
    const float* skipE_W = F(90); const float* skipE_b = F(91);
    const float* e1_W = F(92);    const float* e1_b = F(93);
    const float* ggc_W = F(94);
    const float* gru_wih = F(95); const float* gru_whh = F(96);
    const float* gru_bih = F(97); const float* gru_bhh = F(98);
    const float* e2_W = F(99);    const float* e2_b = F(100);

    float* ws = (float*)d_ws;
    size_t off = 0;
    float* bufA  = ws + off; off += 32768000;   // (B,32,N,64)
    float* bufB  = ws + off; off += 29696000;   // (B,32,N,58) capacity
    float* bufC  = ws + off; off += 29696000;
    float* skip  = ws + off; off += 1024000;    // (B,N,64)
    float* ymat  = ws + off; off += 2048000;    // (B,N,128)
    float* yflat = ws + off; off += 2048000;    // (B,128,N)
    float* hg    = ws + off; off += 2048000;    // (32000,64)
    float* mbuf  = ws + off; off += 2048000;
    float* agg   = ws + off; off += 2048000;
    float* gi    = ws + off; off += 6144000;    // (32000,192)
    float* gh    = ws + off; off += 6144000;
    float* stats = ws + off; off += 32;

    const int TPB = 256;
    auto blocks = [](long long n, int t) { return (unsigned)((n + t - 1) / t); };

    // ---- start_conv, skip0 ----
    { long long tot = 32768000LL;
      k_start<<<blocks(tot, TPB), TPB, 0, stream>>>(x, start_W, start_b, bufA); }
    { long long tot = 1024000LL;
      k_skip_conv<<<blocks(tot, TPB), TPB, 0, stream>>>(x, skip0_W, skip0_b, skip, 12, 64, 0); }

    const int Tins[3]  = {64, 58, 52};
    const int Touts[3] = {58, 52, 46};
    float* layin[3] = {bufA, bufC, bufA};
    float* gated[3] = {bufB, bufB, bufC};
    float* tfout[3] = {bufC, bufA, bufB};

    for (int L = 0; L < 3; ++L) {
        int base = 6 + L * 28;
        int Tin = Tins[L], Tout = Touts[L];
        long long elems = (long long)B_ * 32 * N_ * Tout;
        long long Nn = elems / 16;                 // transformer node count
        IncArgs ia;
        for (int j = 0; j < 4; ++j) {
            ia.fw[j] = F(base + 2 * j);     ia.fb[j] = F(base + 2 * j + 1);
            ia.gw[j] = F(base + 8 + 2 * j); ia.gb[j] = F(base + 8 + 2 * j + 1);
        }
        // gated TCN: h = tanh(inception_f) * sigmoid(inception_g)
        k_inception<<<blocks(elems, TPB), TPB, 0, stream>>>(layin[L], gated[L], ia, Tin, Tout);
        // skip += skip_conv(h)
        k_skip_conv<<<blocks(1024000LL, TPB), TPB, 0, stream>>>(
            gated[L], F(base + 16), F(base + 17), skip, 32, Tout, 1);
        // lin_skip of TransformerConv: tfout = xf @ Ws + bs  (WMMA, K=16)
        wmma_gemm_t<16, false, false><<<(unsigned)(Nn / 16), 32, 0, stream>>>(
            gated[L], 16, F(base + 24), 16, tfout[L], 16, 1, F(base + 25));
        // + residual[..., -Tout:]
        k_res_add<<<blocks(elems, TPB), TPB, 0, stream>>>(tfout[L], layin[L], Tin, Tout);
        // sparse attention on Tout edges
        k_attn<<<1, 64, 0, stream>>>(gated[L], tfout[L], src, dst, Tout,
                                     F(base + 18), F(base + 19), F(base + 20),
                                     F(base + 21), F(base + 22), F(base + 23));
        // layer norm over (C,N,T) per batch
        int perB = 32 * N_ * Tout;
        k_fill<<<1, 32, 0, stream>>>(stats, 0.f, 32);
        dim3 rg(64, B_);
        k_ln_reduce<<<rg, TPB, 0, stream>>>(tfout[L], stats, perB);
        k_ln_apply<<<blocks(elems, TPB), TPB, 0, stream>>>(
            tfout[L], stats, F(base + 26), F(base + 27), perB);
    }

    // ---- skipE (final h = tfout[2] = bufB, T=46) ----
    k_skip_conv<<<blocks(1024000LL, TPB), TPB, 0, stream>>>(
        tfout[2], skipE_W, skipE_b, skip, 32, 46, 1);

    // ---- end_conv_1: ymat(B,N,128) = relu(skip) @ e1_W + e1_b (WMMA, K=64) ----
    wmma_gemm_t<64, false, true><<<(16000 / 16) * (128 / 16), 32, 0, stream>>>(
        skip, 64, e1_W, 128, ymat, 128, 128 / 16, e1_b);
    // (B,N,128) -> (B,128,N) flat order needed by GGC reshape
    k_transpose_y<<<blocks(2048000LL, TPB), TPB, 0, stream>>>(ymat, yflat);
    k_copy<<<blocks(2048000LL, TPB), TPB, 0, stream>>>(yflat, hg, 2048000LL);

    // ---- GatedGraphConv(64, 10, aggr='max') ----
    for (int l = 0; l < 10; ++l) {
        // m = h @ W_l (WMMA)
        wmma_gemm_t<64, false, false><<<(32000 / 16) * (64 / 16), 32, 0, stream>>>(
            hg, 64, ggc_W + (long long)l * 4096, 64, mbuf, 64, 64 / 16, nullptr);
        k_fill<<<blocks(2048000LL, TPB), TPB, 0, stream>>>(agg, -3.0e38f, 2048000LL);
        k_ggc_aggr<<<1, 64, 0, stream>>>(mbuf, src, dst, agg, 1024);
        k_ggc_fix<<<blocks(2048000LL, TPB), TPB, 0, stream>>>(agg, 2048000LL);
        // gi = agg @ wih^T + bih ; gh = h @ whh^T + bhh  (WMMA, transposed B)
        wmma_gemm_t<64, true, false><<<(32000 / 16) * (192 / 16), 32, 0, stream>>>(
            agg, 64, gru_wih, 64, gi, 192, 192 / 16, gru_bih);
        wmma_gemm_t<64, true, false><<<(32000 / 16) * (192 / 16), 32, 0, stream>>>(
            hg, 64, gru_whh, 64, gh, 192, 192 / 16, gru_bhh);
        k_gru<<<blocks(2048000LL, TPB), TPB, 0, stream>>>(hg, gi, gh, 32000LL);
    }

    // ---- end_conv_2: out = relu(y + yf) @ e2_W + e2_b ----
    k_e2<<<blocks(192000LL, TPB), TPB, 0, stream>>>(yflat, hg, e2_W, e2_b, (float*)d_out);
}
